// Qwen35GdnLikeFusableModel_59442347376718
// MI455X (gfx1250) — matrix operand order repaired and checked by reading, not compiled, not run on hardware
//
#include <hip/hip_runtime.h>
#include <hip/hip_fp16.h>
#include <math.h>

typedef __attribute__((ext_vector_type(16))) _Float16 v16h;
typedef __attribute__((ext_vector_type(8)))  _Float16 v8h;
typedef __attribute__((ext_vector_type(8)))  float    v8f;

#define NTOK        32768          // 4 * 8192 tokens
#define K_DIM       2048
#define CHUNK       256            // K elements staged in LDS per step
#define LDS_STRIDE  264            // 256 + 8 halves pad -> row base advances 4 banks
#define N_KTILES    (K_DIM / 32)   // 64
#define WP_HALVES   (8 * N_KTILES * 32 * 16)   // 8 col-tiles * 64 ktiles * 32 lanes * 16 halves

// ---------------- prep kernel 1: pack w_a|w_b into fp16 WMMA-B lane layout --------------
// B lane layout used here (contiguous K): lane<16 -> N=lane,  K = ktile*32 + 0..15 (16 halves)
//                            lane>=16 -> N=lane-16, K = ktile*32 + 16..31
__global__ void pack_weights(const float* __restrict__ wa, const float* __restrict__ wb,
                             _Float16* __restrict__ Wp) {
    int tid  = blockIdx.x * 256 + threadIdx.x;      // [0, WP_HALVES)
    int h    = tid & 15;
    int lane = (tid >> 4) & 31;
    int ktg  = (tid >> 9) & (N_KTILES - 1);
    int ct   = tid >> 15;                           // 0..7 column tile
    int K    = ktg * 32 + ((lane >> 4) << 4) + h;
    int n    = ct * 16 + (lane & 15);               // global column 0..127
    float val = (ct < 4) ? wa[n * K_DIM + K] : wb[(n - 64) * K_DIM + K];
    Wp[tid] = (_Float16)val;
}

// ---------------- prep kernel 2: v[k] = sum_d w_qkv[d,k] + sum_d w_z[d,k] --------------
__global__ void prep_v(const float* __restrict__ wqkv, const float* __restrict__ wz,
                       float* __restrict__ v) {
    int k = blockIdx.x * 256 + threadIdx.x;
    if (k >= K_DIM) return;
    float acc = 0.f;
    for (int d = 0; d < 768; ++d) acc += wqkv[d * K_DIM + k];
    for (int d = 0; d < 512; ++d) acc += wz[d * K_DIM + k];
    v[k] = acc;
}

// ---------------- main kernel: 16 tokens / block, 8 waves = 8 col tiles ----------------
__global__ __launch_bounds__(256) void gdn_main(const float* __restrict__ x,
                                                const _Float16* __restrict__ Wp,
                                                const float* __restrict__ v,
                                                float* __restrict__ out) {
    __shared__ _Float16 sx[16 * LDS_STRIDE];   // fp16 x tile, padded rows
    __shared__ float    sacc[8][16][16];       // per-wave 16x16 f32 results

    const int  t    = threadIdx.x;
    const int  wave = t >> 5;                  // column tile 0..7
    const int  lane = t & 31;
    const int  m_ld = t >> 4;                  // token row owned in load/combine phase
    const int  q    = t & 15;
    const long tok0 = (long)blockIdx.x * 16;
    const float* xrow = x + (tok0 + m_ld) * (long)K_DIM;

    v8f   acc  = {};
    float vdot = 0.f;

    const int mrow = lane & 15;                // A-fragment row
    const int off  = (lane >> 4) << 4;         // 0 or 16 (contiguous K layout, matches packed B)

    for (int c = 0; c < K_DIM / CHUNK; ++c) {
        const int c0 = c * CHUNK;
        if (c + 1 < K_DIM / CHUNK)
            __builtin_prefetch(xrow + c0 + CHUNK, 0, 1);   // global_prefetch_b8

        // ---- load phase: fp32 -> fp16 into LDS, fuse v-dot ----
        #pragma unroll
        for (int i = 0; i < 4; ++i) {
            int f4  = q + 16 * i;              // float4 index within row chunk
            int col = c0 + f4 * 4;
            float4 f = *(const float4*)(xrow + col);
            vdot += f.x * v[col] + f.y * v[col + 1] + f.z * v[col + 2] + f.w * v[col + 3];
            _Float16* d = &sx[m_ld * LDS_STRIDE + f4 * 4];
            d[0] = (_Float16)f.x; d[1] = (_Float16)f.y;
            d[2] = (_Float16)f.z; d[3] = (_Float16)f.w;
        }
        __syncthreads();

        // ---- compute phase: 8 x v_wmma_f32_16x16x32_f16 per wave ----
        #pragma unroll
        for (int kt = 0; kt < CHUNK / 32; ++kt) {
            v8h lo = *(const v8h*)(&sx[mrow * LDS_STRIDE + kt * 32 + off]);
            v8h hi = *(const v8h*)(&sx[mrow * LDS_STRIDE + kt * 32 + off + 8]);
            union { v16h v; v8h h[2]; } au;
            au.h[0] = lo; au.h[1] = hi;

            int ktg = c * (CHUNK / 32) + kt;
            v16h b = *(const v16h*)(Wp + (((wave * N_KTILES) + ktg) * 32 + lane) * 16);

            acc = __builtin_amdgcn_wmma_f32_16x16x32_f16(
                      false, au.v, false, b, (short)0, acc, false, false);
        }
        __syncthreads();
    }

    // ---- spill D tiles: lane L, vgpr r -> token m = r + 8*(L>=16), col = L%16 ----
    #pragma unroll
    for (int r = 0; r < 8; ++r) {
        int m = r + ((lane >> 4) << 3);
        sacc[wave][m][lane & 15] = acc[r];
    }
    __syncthreads();

    // ---- combine: out[tok] = x.v + sum_h a_h * sigmoid(b_h) ----
    float s = vdot;
    #pragma unroll
    for (int u = 0; u < 4; ++u) {
        int hd = q + u * 16;                               // head 0..63, each once
        float av = sacc[hd >> 4][m_ld][hd & 15];           // a columns: tiles 0..3
        float bv = sacc[4 + (hd >> 4)][m_ld][hd & 15];     // b columns: tiles 4..7
        s += av * (1.0f / (1.0f + __expf(-bv)));
    }
    // reduce across the 16 threads sharing token m_ld (xor<16 stays in group)
    #pragma unroll
    for (int k = 8; k >= 1; k >>= 1) s += __shfl_xor(s, k, 32);
    if (q == 0) out[tok0 + m_ld] = s;
}

extern "C" void kernel_launch(void* const* d_in, const int* in_sizes, int n_in,
                              void* d_out, int out_size, void* d_ws, size_t ws_size,
                              hipStream_t stream) {
    const float* x    = (const float*)d_in[0];
    const float* wqkv = (const float*)d_in[1];
    const float* wz   = (const float*)d_in[2];
    const float* wb   = (const float*)d_in[3];
    const float* wa   = (const float*)d_in[4];

    _Float16* Wp = (_Float16*)d_ws;
    float*    v  = (float*)((char*)d_ws + (size_t)WP_HALVES * sizeof(_Float16));
    float*    out = (float*)d_out;

    pack_weights<<<WP_HALVES / 256, 256, 0, stream>>>(wa, wb, Wp);
    prep_v<<<(K_DIM + 255) / 256, 256, 0, stream>>>(wqkv, wz, v);
    gdn_main<<<NTOK / 16, 256, 0, stream>>>(x, Wp, v, out);
}
